// MultiHeadSelfAttention_91018946937275
// MI455X (gfx1250) — matrix operand order repaired and checked by reading, not compiled, not run on hardware
//
#include <hip/hip_runtime.h>

#define EMBED  1024
#define NHEADS 4
#define HDIM   256
#define BATCH  4
#define SEQ    2048
#define NROWS  (BATCH * SEQ)   // 8192

typedef __attribute__((ext_vector_type(16))) __bf16 bf16x16;
typedef __attribute__((ext_vector_type(8)))  float  f32x8;
typedef int v4i __attribute__((vector_size(16)));

#define AS1 __attribute__((address_space(1)))
#define AS3 __attribute__((address_space(3)))

// ---- gfx1250 async global->LDS copy (ASYNCcnt path), with sync fallback ----
#if defined(__has_builtin)
#if __has_builtin(__builtin_amdgcn_global_load_async_to_lds_b128)
#define HAVE_ASYNC_LDS 1
#endif
#endif
#ifndef HAVE_ASYNC_LDS
#define HAVE_ASYNC_LDS 0
#endif

__device__ __forceinline__ void async_copy_b128(void* lds_dst, const void* gsrc) {
#if HAVE_ASYNC_LDS
  // Flat LDS addresses carry the LDS byte offset in the low 32 bits (ISA 10.2).
  __builtin_amdgcn_global_load_async_to_lds_b128(
      (AS1 v4i*)(unsigned long long)gsrc,
      (AS3 v4i*)(unsigned int)(unsigned long long)lds_dst, 0, 0);
#else
  *(uint4*)lds_dst = *(const uint4*)gsrc;
#endif
}

#if HAVE_ASYNC_LDS
#if defined(__has_builtin) && __has_builtin(__builtin_amdgcn_s_wait_asynccnt)
#define WAIT_ASYNC_ALL() __builtin_amdgcn_s_wait_asynccnt(0)
#else
#define WAIT_ASYNC_ALL() asm volatile("s_wait_asynccnt 0" ::: "memory")
#endif
#else
#define WAIT_ASYNC_ALL() do {} while (0)
#endif

union Frag16 {
  bf16x16      v;
  unsigned int u[8];
};

__device__ __forceinline__ f32x8 wmma_bf16(const Frag16& a, const Frag16& b, f32x8 c) {
  return __builtin_amdgcn_wmma_f32_16x16x32_bf16(false, a.v, false, b.v, (short)0, c,
                                                 false, false);
}

__device__ __forceinline__ unsigned short f32_to_bf16(float f) {
  unsigned int u = __builtin_bit_cast(unsigned int, f);
  u += 0x7fffu + ((u >> 16) & 1u);   // round-to-nearest-even
  return (unsigned short)(u >> 16);
}

// A-matrix (16-bit, 16x32): VGPR j holds K, K+1 at K = (j<4?0:16) + 8*half + 2*(j&3)
__device__ __forceinline__ int kbaseA(int j, int h) {
  return ((j < 4) ? 0 : 16) + 8 * h + 2 * (j & 3);
}
// B-matrix (16-bit, 32x16): VGPR j holds K, K+1 at K = (j<4?0:16) + 8*half + 2*(j&3)
__device__ __forceinline__ int kbaseB(int j, int h) { return ((j < 4) ? 0 : 16) + 8 * h + 2 * (j & 3); }

// ---------------------------------------------------------------- converts
__global__ void cvt_f32_to_bf16(const float* __restrict__ src,
                                unsigned short* __restrict__ dst, int n) {
  int i = blockIdx.x * blockDim.x + threadIdx.x;
  if (i < n) dst[i] = f32_to_bf16(src[i]);
}

// dst[n*K + k] = bf16(src[k*N + n])
__global__ void trans_cvt_bf16(const float* __restrict__ src,
                               unsigned short* __restrict__ dst, int K, int N) {
  int i = blockIdx.x * blockDim.x + threadIdx.x;
  if (i < K * N) {
    int k = i % K;
    int n = i / K;
    dst[i] = f32_to_bf16(src[(long)k * N + n]);
  }
}

// ---------------------------------------------------------------- GEMM
// C[M,Ncols] = A[M,1024] * Bt[Ncols,1024]^T + bias. 64x64 tiles, BK=32, double-buffered.
// MODE 0: scatter to q[B,H,N,hd], k[B,H,N,hd], v^T[B,H,hd,N] (bf16).
// MODE 1: add resid, write f32.
template <int MODE>
__global__ __launch_bounds__(128) void gemm64(
    const unsigned short* __restrict__ A, const unsigned short* __restrict__ Bt,
    const float* __restrict__ bias, const float* __restrict__ resid,
    unsigned short* __restrict__ qb, unsigned short* __restrict__ kb,
    unsigned short* __restrict__ vbt, float* __restrict__ yout) {
  __shared__ __align__(16) unsigned short lA[2][64 * 40];
  __shared__ __align__(16) unsigned short lB[2][64 * 40];

  const int m0 = blockIdx.y * 64;
  const int n0 = blockIdx.x * 64;
  const int tid = threadIdx.x;
  const int wave = tid >> 5;
  const int lane = tid & 31;
  const int lm = lane & 15;
  const int lh = lane >> 4;

  auto stage = [&](int buf, int k0) {
#pragma unroll
    for (int it = 0; it < 2; ++it) {
      int ch = tid + it * 128;   // 0..255
      int row = ch >> 2;
      int cg = ch & 3;
      async_copy_b128(&lA[buf][row * 40 + cg * 8],
                      A + (long)(m0 + row) * 1024 + k0 + cg * 8);
      async_copy_b128(&lB[buf][row * 40 + cg * 8],
                      Bt + (long)(n0 + row) * 1024 + k0 + cg * 8);
    }
  };

  f32x8 zero8 = {0.f, 0.f, 0.f, 0.f, 0.f, 0.f, 0.f, 0.f};
  f32x8 acc[4] = {zero8, zero8, zero8, zero8};

  stage(0, 0);
  for (int i = 0; i < 32; ++i) {
    WAIT_ASYNC_ALL();
    __syncthreads();
    if (i + 1 < 32) stage((i + 1) & 1, (i + 1) * 32);
    const unsigned short* la = lA[i & 1];
    const unsigned short* lb = lB[i & 1];

    // load A fragment + all 4 B fragments up front so DS loads pipeline,
    // then issue the 4 independent-accumulator WMMAs.
    Frag16 af, bfs[4];
#pragma unroll
    for (int j = 0; j < 8; ++j)
      af.u[j] = *(const unsigned int*)&la[(wave * 16 + lm) * 40 + kbaseA(j, lh)];
#pragma unroll
    for (int c = 0; c < 4; ++c)
#pragma unroll
      for (int j = 0; j < 8; ++j)
        bfs[c].u[j] = *(const unsigned int*)&lb[(c * 16 + lm) * 40 + kbaseB(j, lh)];
#pragma unroll
    for (int c = 0; c < 4; ++c) acc[c] = wmma_bf16(af, bfs[c], acc[c]);
    __syncthreads();
  }

#pragma unroll
  for (int c = 0; c < 4; ++c) {
#pragma unroll
    for (int r = 0; r < 8; ++r) {
      int row = m0 + wave * 16 + r + 8 * lh;
      int col = n0 + c * 16 + lm;
      float val = acc[c][r] + bias[col];
      if (MODE == 0) {
        int which = col >> 10;   // 0=q 1=k 2=v
        int cc = col & 1023;
        int head = cc >> 8;
        int d = cc & 255;
        int b = row >> 11;
        int nn = row & 2047;
        unsigned short bv = f32_to_bf16(val);
        if (which == 0)
          qb[((long)(b * NHEADS + head) * SEQ + nn) * HDIM + d] = bv;
        else if (which == 1)
          kb[((long)(b * NHEADS + head) * SEQ + nn) * HDIM + d] = bv;
        else
          vbt[((long)(b * NHEADS + head) * HDIM + d) * SEQ + nn] = bv;  // V^T
      } else {
        val += resid[(long)row * EMBED + col];
        yout[(long)row * EMBED + col] = val;
      }
    }
  }
}

// ---------------------------------------------------------------- flash attention
// grid (SEQ/64, BATCH*NHEADS), 128 threads; wave owns 16 query rows; kj tiles of 32,
// K and V^T tiles double-buffered via async LDS copies.
__global__ __launch_bounds__(128) void attn_kernel(
    const unsigned short* __restrict__ q, const unsigned short* __restrict__ k,
    const unsigned short* __restrict__ vt, unsigned short* __restrict__ ob) {
  __shared__ __align__(16) unsigned short lK[2][32 * 264];   // [kj][d]
  __shared__ __align__(16) unsigned short lVT[2][256 * 40];  // [d][kj]
  __shared__ __align__(16) unsigned short lP[64 * 40];       // per-wave P rows

  const int bh = blockIdx.y;
  const int q0 = blockIdx.x * 64;
  const int tid = threadIdx.x;
  const int wave = tid >> 5;
  const int lane = tid & 31;
  const int lm = lane & 15;
  const int lh = lane >> 4;

  const unsigned short* qbh = q + (long)bh * SEQ * HDIM;
  const unsigned short* kbh = k + (long)bh * SEQ * HDIM;
  const unsigned short* vtbh = vt + (long)bh * HDIM * SEQ;

  auto stage = [&](int buf, int kt) {
#pragma unroll
    for (int it = 0; it < 8; ++it) {
      int ch = tid + it * 128;   // 0..1023
      int kr = ch >> 5, kc = ch & 31;
      async_copy_b128(&lK[buf][kr * 264 + kc * 8],
                      kbh + (long)(kt + kr) * HDIM + kc * 8);
      int vr = ch >> 2, vc = ch & 3;
      async_copy_b128(&lVT[buf][vr * 40 + vc * 8],
                      vtbh + (long)vr * SEQ + kt + vc * 8);
    }
  };

  // Q tile (16 rows x 256) as 8 persistent A-fragments
  Frag16 qf[8];
  {
    const int qrow = q0 + wave * 16 + lm;
    const unsigned int* qrp = (const unsigned int*)(qbh + (long)qrow * HDIM);
#pragma unroll
    for (int t = 0; t < 8; ++t)
#pragma unroll
      for (int j = 0; j < 8; ++j)
        qf[t].u[j] = qrp[(32 * t + kbaseA(j, lh)) >> 1];
  }

  f32x8 zero8 = {0.f, 0.f, 0.f, 0.f, 0.f, 0.f, 0.f, 0.f};
  f32x8 o[16];
#pragma unroll
  for (int f = 0; f < 16; ++f) o[f] = zero8;
  float row_m[8], row_l[8];
#pragma unroll
  for (int r = 0; r < 8; ++r) { row_m[r] = -3.0e38f; row_l[r] = 0.f; }

  const float scale = 0.0625f;   // 1/sqrt(256)

  stage(0, 0);
  for (int i = 0; i < SEQ / 32; ++i) {
    WAIT_ASYNC_ALL();
    __syncthreads();
    if (i + 1 < SEQ / 32) stage((i + 1) & 1, (i + 1) * 32);
    const unsigned short* lk = lK[i & 1];
    const unsigned short* lv = lVT[i & 1];

    // scores: two 16x16 subtiles; preload all 8 B fragments per subtile so the
    // 16 ds_load_b128 pipeline ahead of the dependent WMMA accumulate chain.
    f32x8 s[2];
#pragma unroll
    for (int c = 0; c < 2; ++c) {
      Frag16 bfs[8];
#pragma unroll
      for (int t = 0; t < 8; ++t)
#pragma unroll
        for (int j = 0; j < 8; ++j)
          bfs[t].u[j] =
              *(const unsigned int*)&lk[(c * 16 + lm) * 264 + 32 * t + kbaseB(j, lh)];
      s[c] = zero8;
#pragma unroll
      for (int t = 0; t < 8; ++t) s[c] = wmma_bf16(qf[t], bfs[t], s[c]);
    }

    // online softmax (row = register index r + 8*half; column = lane within half)
    float a_scale[8];
#pragma unroll
    for (int r = 0; r < 8; ++r) {
      float s0 = s[0][r] * scale;
      float s1 = s[1][r] * scale;
      float mx = fmaxf(s0, s1);
#pragma unroll
      for (int off = 1; off < 16; off <<= 1) mx = fmaxf(mx, __shfl_xor(mx, off, 32));
      float nm = fmaxf(row_m[r], mx);
      float alpha = __expf(row_m[r] - nm);
      float p0 = __expf(s0 - nm);
      float p1 = __expf(s1 - nm);
      float sum = p0 + p1;
#pragma unroll
      for (int off = 1; off < 16; off <<= 1) sum += __shfl_xor(sum, off, 32);
      row_l[r] = row_l[r] * alpha + sum;
      row_m[r] = nm;
      a_scale[r] = alpha;
      int prow = wave * 16 + r + 8 * lh;
      lP[prow * 40 + lm] = f32_to_bf16(p0);
      lP[prow * 40 + 16 + lm] = f32_to_bf16(p1);
    }

#pragma unroll
    for (int f = 0; f < 16; ++f)
#pragma unroll
      for (int r = 0; r < 8; ++r) o[f][r] *= a_scale[r];

    __syncthreads();   // lP ordering (uniform across waves)

    // O += P(16x32) * V(32x16 per d-subtile): groups of 4 independent WMMAs
    Frag16 pf;
#pragma unroll
    for (int j = 0; j < 8; ++j)
      pf.u[j] = *(const unsigned int*)&lP[(wave * 16 + lm) * 40 + kbaseA(j, lh)];
#pragma unroll
    for (int g = 0; g < 4; ++g) {
      Frag16 bfs[4];
#pragma unroll
      for (int f4 = 0; f4 < 4; ++f4)
#pragma unroll
        for (int j = 0; j < 8; ++j)
          bfs[f4].u[j] =
              *(const unsigned int*)&lv[((g * 4 + f4) * 16 + lm) * 40 + kbaseB(j, lh)];
#pragma unroll
      for (int f4 = 0; f4 < 4; ++f4)
        o[g * 4 + f4] = wmma_bf16(pf, bfs[f4], o[g * 4 + f4]);
    }
    __syncthreads();
  }

  const int b = bh >> 2;
  const int head = bh & 3;
#pragma unroll
  for (int r = 0; r < 8; ++r) {
    float inv = 1.0f / row_l[r];
    int row = q0 + wave * 16 + r + 8 * lh;
    unsigned short* orow = ob + ((long)(b * SEQ + row)) * EMBED + head * HDIM;
#pragma unroll
    for (int f = 0; f < 16; ++f) orow[f * 16 + lm] = f32_to_bf16(o[f][r] * inv);
  }
}

// ---------------------------------------------------------------- layernorm
__global__ __launch_bounds__(256) void ln_kernel(const float* __restrict__ y,
                                                 const float* __restrict__ gamma,
                                                 const float* __restrict__ beta,
                                                 float* __restrict__ out) {
  __shared__ float rs[8], rs2[8];
  long row = blockIdx.x;
  const float* yr = y + row * EMBED;
  float s = 0.f, s2 = 0.f;
  for (int i = threadIdx.x; i < EMBED; i += 256) {
    float t = yr[i];
    s += t;
    s2 += t * t;
  }
#pragma unroll
  for (int off = 16; off; off >>= 1) {
    s += __shfl_down(s, off, 32);
    s2 += __shfl_down(s2, off, 32);
  }
  if ((threadIdx.x & 31) == 0) {
    rs[threadIdx.x >> 5] = s;
    rs2[threadIdx.x >> 5] = s2;
  }
  __syncthreads();
  if (threadIdx.x == 0) {
    float a = 0.f, b = 0.f;
#pragma unroll
    for (int i = 0; i < 8; ++i) { a += rs[i]; b += rs2[i]; }
    rs[0] = a;
    rs2[0] = b;
  }
  __syncthreads();
  float mean = rs[0] * (1.0f / EMBED);
  float var = rs2[0] * (1.0f / EMBED) - mean * mean;
  float inv = rsqrtf(var + 1e-5f);
  for (int i = threadIdx.x; i < EMBED; i += 256)
    out[row * EMBED + i] = (yr[i] - mean) * inv * gamma[i] + beta[i];
}

// ---------------------------------------------------------------- launch
extern "C" void kernel_launch(void* const* d_in, const int* in_sizes, int n_in,
                              void* d_out, int out_size, void* d_ws, size_t ws_size,
                              hipStream_t stream) {
  (void)in_sizes; (void)n_in; (void)out_size; (void)ws_size;
  const float* x        = (const float*)d_in[0];
  const float* w_qkv    = (const float*)d_in[1];
  const float* b_qkv    = (const float*)d_in[2];
  const float* w_out    = (const float*)d_in[3];
  const float* b_out    = (const float*)d_in[4];
  const float* ln_gamma = (const float*)d_in[5];
  const float* ln_beta  = (const float*)d_in[6];
  float* out = (float*)d_out;

  size_t off = 0;
  auto alloc = [&](size_t bytes) {
    void* p = (char*)d_ws + off;
    off += (bytes + 255) & ~(size_t)255;
    return p;
  };
  unsigned short* xb    = (unsigned short*)alloc((size_t)NROWS * EMBED * 2);
  unsigned short* wqkvT = (unsigned short*)alloc((size_t)3 * EMBED * EMBED * 2);
  unsigned short* woutT = (unsigned short*)alloc((size_t)EMBED * EMBED * 2);
  unsigned short* qb    = (unsigned short*)alloc((size_t)BATCH * NHEADS * SEQ * HDIM * 2);
  unsigned short* kb    = (unsigned short*)alloc((size_t)BATCH * NHEADS * SEQ * HDIM * 2);
  unsigned short* vbt   = (unsigned short*)alloc((size_t)BATCH * NHEADS * SEQ * HDIM * 2);
  unsigned short* obb   = (unsigned short*)alloc((size_t)NROWS * EMBED * 2);
  float*          yf    = (float*)alloc((size_t)NROWS * EMBED * 4);

  {
    int n = NROWS * EMBED;
    cvt_f32_to_bf16<<<(n + 255) / 256, 256, 0, stream>>>(x, xb, n);
  }
  {
    int n = EMBED * 3 * EMBED;
    trans_cvt_bf16<<<(n + 255) / 256, 256, 0, stream>>>(w_qkv, wqkvT, EMBED, 3 * EMBED);
  }
  {
    int n = EMBED * EMBED;
    trans_cvt_bf16<<<(n + 255) / 256, 256, 0, stream>>>(w_out, woutT, EMBED, EMBED);
  }

  gemm64<0><<<dim3(3 * EMBED / 64, NROWS / 64), 128, 0, stream>>>(
      xb, wqkvT, b_qkv, nullptr, qb, kb, vbt, nullptr);

  attn_kernel<<<dim3(SEQ / 64, BATCH * NHEADS), 128, 0, stream>>>(qb, kb, vbt, obb);

  gemm64<1><<<dim3(EMBED / 64, NROWS / 64), 128, 0, stream>>>(
      obb, woutT, b_out, x, nullptr, nullptr, nullptr, yf);

  ln_kernel<<<NROWS, 256, 0, stream>>>(yf, ln_gamma, ln_beta, out);
}
